// CustomStellarEncoder_2_51342039056842
// MI455X (gfx1250) — matrix operand order repaired and checked by reading, not compiled, not run on hardware
//
#include <hip/hip_runtime.h>
#include <hip/hip_bf16.h>

typedef __attribute__((ext_vector_type(16))) _Float16 v16h;
typedef __attribute__((ext_vector_type(8)))  _Float16 h8;
typedef __attribute__((ext_vector_type(8)))  float    v8f;

#define DH 128
#define A_STRIDE 56   // halves per A-tile row in LDS (16B aligned rows, conflict-free)
#define EPS 1e-5f

// ---------------------------------------------------------------------------
// Utility kernels
// ---------------------------------------------------------------------------
__global__ void sage_zero_kernel(float* __restrict__ p, size_t n) {
    size_t i = (size_t)blockIdx.x * blockDim.x + threadIdx.x;
    size_t stride = (size_t)gridDim.x * blockDim.x;
    for (; i < n; i += stride) p[i] = 0.0f;
}

// Convert W [K,128] f32 (row-major) -> WT [128][K+8] f16 (column-major, padded)
__global__ void sage_convert_wt_kernel(const float* __restrict__ W,
                                       _Float16* __restrict__ WT, int K) {
    int i = blockIdx.x * blockDim.x + threadIdx.x;
    if (i >= K * DH) return;
    int k = i >> 7, c = i & (DH - 1);
    WT[(size_t)c * (K + 8) + k] = (_Float16)W[i];
}

// ---------------------------------------------------------------------------
// WMMA GEMM: C[N,128] = A[N,K] @ W[K,128] (+ bias) (+= if accumulate)
// block = 128 threads (4 waves); each wave computes 32 rows x 128 cols
// (two A fragments per wave so every B fragment feeds two WMMAs)
// ---------------------------------------------------------------------------
extern __shared__ _Float16 g_lds[];

__global__ __launch_bounds__(128)
void sage_gemm_kernel(const float* __restrict__ A, const _Float16* __restrict__ WT,
                      const float* __restrict__ bias, float* __restrict__ C,
                      int N, int K, int accumulate)
{
    const int wstride = K + 8;
    _Float16* wt    = g_lds;                       // 128 * wstride halves
    _Float16* atile = g_lds + 128 * wstride;       // 4 waves * 32 rows * A_STRIDE halves

    const int tid  = threadIdx.x;
    const int wave = tid >> 5;
    const int lane = tid & 31;
    const int m    = lane & 15;
    const int hh   = lane >> 4;

    // cooperative copy of f16 weights into LDS (dword granularity)
    {
        const unsigned* src = reinterpret_cast<const unsigned*>(WT);
        unsigned* dst = reinterpret_cast<unsigned*>(wt);
        const int nd = 128 * wstride / 2;
        for (int i = tid; i < nd; i += 128) dst[i] = src[i];
    }
    __syncthreads();

    const int row0 = blockIdx.x * 128 + wave * 32;     // 32 rows per wave
    _Float16* aw = atile + wave * 32 * A_STRIDE;

    // A staging assignment: lane -> (row = lane/2, 16-col half = lane&1), x2 strips
    const int srow = lane >> 1;
    const int scol = (lane & 1) << 4;
    const int arow0 = row0 + srow;
    const int arow1 = row0 + 16 + srow;
    const bool av0 = (arow0 < N);
    const bool av1 = (arow1 < N);
    const float4* ag0 = reinterpret_cast<const float4*>(A + (size_t)arow0 * K + scol);
    const float4* ag1 = reinterpret_cast<const float4*>(A + (size_t)arow1 * K + scol);
    _Float16* sp0 = aw + srow * A_STRIDE + scol;
    _Float16* sp1 = aw + (16 + srow) * A_STRIDE + scol;

    v8f acc0[8], acc1[8];
#pragma unroll
    for (int t = 0; t < 8; ++t) { acc0[t] = (v8f)(0.0f); acc1[t] = (v8f)(0.0f); }

    for (int kb = 0; kb < K; kb += 32) {
        const int b = kb >> 2;
        // ---- stage A: 2 strips x 16 rows x 32 halves ----
        float4 q0 = {}, q1 = {}, q2 = {}, q3 = {};
        if (av0) { q0 = ag0[b+0]; q1 = ag0[b+1]; q2 = ag0[b+2]; q3 = ag0[b+3]; }
        float4 r0 = {}, r1 = {}, r2 = {}, r3 = {};
        if (av1) { r0 = ag1[b+0]; r1 = ag1[b+1]; r2 = ag1[b+2]; r3 = ag1[b+3]; }

        h8 p0, p1, p2, p3;
        p0[0]=(_Float16)q0.x; p0[1]=(_Float16)q0.y; p0[2]=(_Float16)q0.z; p0[3]=(_Float16)q0.w;
        p0[4]=(_Float16)q1.x; p0[5]=(_Float16)q1.y; p0[6]=(_Float16)q1.z; p0[7]=(_Float16)q1.w;
        p1[0]=(_Float16)q2.x; p1[1]=(_Float16)q2.y; p1[2]=(_Float16)q2.z; p1[3]=(_Float16)q2.w;
        p1[4]=(_Float16)q3.x; p1[5]=(_Float16)q3.y; p1[6]=(_Float16)q3.z; p1[7]=(_Float16)q3.w;
        p2[0]=(_Float16)r0.x; p2[1]=(_Float16)r0.y; p2[2]=(_Float16)r0.z; p2[3]=(_Float16)r0.w;
        p2[4]=(_Float16)r1.x; p2[5]=(_Float16)r1.y; p2[6]=(_Float16)r1.z; p2[7]=(_Float16)r1.w;
        p3[0]=(_Float16)r2.x; p3[1]=(_Float16)r2.y; p3[2]=(_Float16)r2.z; p3[3]=(_Float16)r2.w;
        p3[4]=(_Float16)r3.x; p3[5]=(_Float16)r3.y; p3[6]=(_Float16)r3.z; p3[7]=(_Float16)r3.w;
        *reinterpret_cast<h8*>(sp0)     = p0;
        *reinterpret_cast<h8*>(sp0 + 8) = p1;
        *reinterpret_cast<h8*>(sp1)     = p2;
        *reinterpret_cast<h8*>(sp1 + 8) = p3;

        // ---- A fragments: a[0..7]=K(kb+8h..+7), a[8..15]=K(kb+16+8h..+7) ----
        v16h af0, af1;
        {
            h8 a0 = *reinterpret_cast<const h8*>(&aw[m * A_STRIDE + 8 * hh]);
            h8 a1 = *reinterpret_cast<const h8*>(&aw[m * A_STRIDE + 16 + 8 * hh]);
            h8 a2 = *reinterpret_cast<const h8*>(&aw[(16 + m) * A_STRIDE + 8 * hh]);
            h8 a3 = *reinterpret_cast<const h8*>(&aw[(16 + m) * A_STRIDE + 16 + 8 * hh]);
#pragma unroll
            for (int i = 0; i < 8; ++i) {
                af0[i] = a0[i]; af0[8 + i] = a1[i];
                af1[i] = a2[i]; af1[8 + i] = a3[i];
            }
        }

        // ---- 8 column tiles; each B fragment feeds two WMMAs ----
#pragma unroll
        for (int ct = 0; ct < 8; ++ct) {
            const int col = ct * 16 + m;
            const h8* q = reinterpret_cast<const h8*>(&wt[(size_t)col * wstride + kb + 8 * hh]);
            h8 b0 = q[0], b1 = q[2];
            v16h bfrag;
#pragma unroll
            for (int i = 0; i < 8; ++i) { bfrag[i] = b0[i]; bfrag[8 + i] = b1[i]; }
            acc0[ct] = __builtin_amdgcn_wmma_f32_16x16x32_f16(
                false, af0, false, bfrag, (short)0, acc0[ct], false, false);
            acc1[ct] = __builtin_amdgcn_wmma_f32_16x16x32_f16(
                false, af1, false, bfrag, (short)0, acc1[ct], false, false);
        }
    }

    // ---- store: accS[ct][r] -> row (row0 + 16*S + r + 8h), col (ct*16 + m) ----
    float bv[8];
#pragma unroll
    for (int ct = 0; ct < 8; ++ct) bv[ct] = bias ? bias[ct * 16 + m] : 0.0f;

#pragma unroll
    for (int s = 0; s < 2; ++s) {
        const int rb = row0 + 16 * s;
        const v8f* acc = s ? acc1 : acc0;
        if (rb + 16 <= N) {            // wave-uniform fast path
#pragma unroll
            for (int ct = 0; ct < 8; ++ct) {
                const int col = ct * 16 + m;
#pragma unroll
                for (int r = 0; r < 8; ++r) {
                    size_t o = (size_t)(rb + r + 8 * hh) * DH + col;
                    float v = acc[ct][r] + bv[ct];
                    if (accumulate) v += C[o];
                    C[o] = v;
                }
            }
        } else if (rb < N) {           // ragged tail
#pragma unroll
            for (int ct = 0; ct < 8; ++ct) {
                const int col = ct * 16 + m;
#pragma unroll
                for (int r = 0; r < 8; ++r) {
                    int grow = rb + r + 8 * hh;
                    if (grow < N) {
                        size_t o = (size_t)grow * DH + col;
                        float v = acc[ct][r] + bv[ct];
                        if (accumulate) v += C[o];
                        C[o] = v;
                    }
                }
            }
        }
    }
}

// ---------------------------------------------------------------------------
// BatchNorm stats: st[c] = sum, st[128+c] = sumsq over rows
// ---------------------------------------------------------------------------
__global__ void sage_stats_kernel(const float* __restrict__ h, float* __restrict__ st, int N) {
    const int c = threadIdx.x & (DH - 1);
    const int half = threadIdx.x >> 7;
    float s = 0.0f, s2 = 0.0f;
    for (int i = blockIdx.x * 2 + half; i < N; i += gridDim.x * 2) {
        float v = h[(size_t)i * DH + c];
        s += v; s2 += v * v;
    }
    atomicAdd(&st[c], s);
    atomicAdd(&st[DH + c], s2);
}

__global__ void sage_bn_kernel(const float* __restrict__ h, const float* __restrict__ st,
                               const float* __restrict__ g, const float* __restrict__ be,
                               float* __restrict__ y, size_t total, float invN, int relu) {
    size_t i = (size_t)blockIdx.x * blockDim.x + threadIdx.x;
    if (i >= total) return;
    int c = (int)(i & (DH - 1));
    float mean = st[c] * invN;
    float var  = st[DH + c] * invN - mean * mean;
    float v = (h[i] - mean) * rsqrtf(var + EPS) * g[c] + be[c];
    if (relu) v = fmaxf(v, 0.0f);
    y[i] = v;
}

// ---------------------------------------------------------------------------
// SAGE scatter: one wave per edge; agg[dst] += feat[src]; cnt[dst] += 1
// ---------------------------------------------------------------------------
__global__ void sage_scatter_kernel(const float* __restrict__ feat,
                                    const long long* __restrict__ src,
                                    const long long* __restrict__ dst,
                                    float* __restrict__ agg, float* __restrict__ cnt, int E) {
    int w = blockIdx.x * 8 + (threadIdx.x >> 5);
    if (w >= E) return;
    int lane = threadIdx.x & 31;
    long long s = src[w], d = dst[w];
    const float* fr = feat + (size_t)s * DH;
    float* ar = agg + (size_t)d * DH;
#pragma unroll
    for (int j = 0; j < 4; ++j)
        atomicAdd(&ar[lane + 32 * j], fr[lane + 32 * j]);
    if (lane == 0) atomicAdd(&cnt[d], 1.0f);
}

__global__ void sage_meandiv_kernel(float* __restrict__ agg, const float* __restrict__ cnt,
                                    size_t total) {
    size_t i = (size_t)blockIdx.x * blockDim.x + threadIdx.x;
    if (i >= total) return;
    agg[i] *= 1.0f / fmaxf(cnt[i >> 7], 1.0f);
}

// ---------------------------------------------------------------------------
// Host orchestration
// ---------------------------------------------------------------------------
extern "C" void kernel_launch(void* const* d_in, const int* in_sizes, int n_in,
                              void* d_out, int out_size, void* d_ws, size_t ws_size,
                              hipStream_t stream) {
    (void)n_in; (void)out_size; (void)ws_size;
    const float* x        = (const float*)d_in[0];
    const long long* ei   = (const long long*)d_in[1];
    const float* W_in = (const float*)d_in[2],  *b_in = (const float*)d_in[3];
    const float* g_in = (const float*)d_in[4],  *be_in = (const float*)d_in[5];
    const float* W_h  = (const float*)d_in[6],  *b_h  = (const float*)d_in[7];
    const float* g_h  = (const float*)d_in[8],  *be_h = (const float*)d_in[9];
    const float* Wl1  = (const float*)d_in[10], *bl1  = (const float*)d_in[11];
    const float* Wr1  = (const float*)d_in[12];
    const float* g1   = (const float*)d_in[13], *be1  = (const float*)d_in[14];
    const float* Wl2  = (const float*)d_in[15], *bl2  = (const float*)d_in[16];
    const float* Wr2  = (const float*)d_in[17];
    const float* g2   = (const float*)d_in[18], *be2  = (const float*)d_in[19];

    const int DIN = 256;
    const int N = in_sizes[0] / DIN;       // 50000
    const int E = in_sizes[1] / 2;         // 600000
    const size_t NF = (size_t)N * DH;
    const long long* e_src = ei;
    const long long* e_dst = ei + E;

    float* feat = (float*)d_out;           // output 0: [N,128]
    float* out1 = feat + NF;               // output 1: [N,128]

    // workspace carve (256B aligned)
    char* base = (char*)d_ws;
    size_t off = 0;
    auto carve = [&](size_t bytes) -> char* {
        off = (off + 255) & ~(size_t)255;
        char* p = base + off; off += bytes; return p;
    };
    _Float16* WT_in = (_Float16*)carve((size_t)DH * (DIN + 8) * 2);
    _Float16* WT_h  = (_Float16*)carve((size_t)DH * (DH + 8) * 2);
    _Float16* WT_l1 = (_Float16*)carve((size_t)DH * (DH + 8) * 2);
    _Float16* WT_r1 = (_Float16*)carve((size_t)DH * (DH + 8) * 2);
    _Float16* WT_l2 = (_Float16*)carve((size_t)DH * (DH + 8) * 2);
    _Float16* WT_r2 = (_Float16*)carve((size_t)DH * (DH + 8) * 2);
    float* buf0 = (float*)carve(NF * 4);   // pre-BN linear outputs
    float* buf1 = (float*)carve(NF * 4);   // feat1 temp / neighbor agg
    float* cnt  = (float*)carve((size_t)N * 4);
    float* st   = (float*)carve(2 * DH * 4);

    const int ZB = 2048;                   // zero-kernel grid
    const dim3 cvtIn((DIN * DH + 255) / 256), cvtH((DH * DH + 255) / 256);
    const int gemmGrid = (N + 127) / 128;  // 128 rows per block
    const size_t smemIn = ((size_t)DH * (DIN + 8) + 4 * 32 * A_STRIDE) * 2;
    const size_t smemH  = ((size_t)DH * (DH + 8) + 4 * 32 * A_STRIDE) * 2;
    const int bnGrid = (int)((NF + 255) / 256);
    const float invN = 1.0f / (float)N;
    const int scatGrid = (E + 7) / 8;

    // --- weight conversion (f32 -> f16, column-major padded) ---
    sage_convert_wt_kernel<<<cvtIn, 256, 0, stream>>>(W_in, WT_in, DIN);
    sage_convert_wt_kernel<<<cvtH, 256, 0, stream>>>(W_h,  WT_h,  DH);
    sage_convert_wt_kernel<<<cvtH, 256, 0, stream>>>(Wl1,  WT_l1, DH);
    sage_convert_wt_kernel<<<cvtH, 256, 0, stream>>>(Wr1,  WT_r1, DH);
    sage_convert_wt_kernel<<<cvtH, 256, 0, stream>>>(Wl2,  WT_l2, DH);
    sage_convert_wt_kernel<<<cvtH, 256, 0, stream>>>(Wr2,  WT_r2, DH);

    // --- layer 1: relu(BN(x @ W_in + b_in)) -> buf1 ---
    sage_gemm_kernel<<<gemmGrid, 128, smemIn, stream>>>(x, WT_in, b_in, buf0, N, DIN, 0);
    sage_zero_kernel<<<1, 256, 0, stream>>>(st, 2 * DH);
    sage_stats_kernel<<<512, 256, 0, stream>>>(buf0, st, N);
    sage_bn_kernel<<<bnGrid, 256, 0, stream>>>(buf0, st, g_in, be_in, buf1, NF, invN, 1);

    // --- layer 2: relu(BN(h @ W_h + b_h)) -> feat (d_out first half) ---
    sage_gemm_kernel<<<gemmGrid, 128, smemH, stream>>>(buf1, WT_h, b_h, buf0, N, DH, 0);
    sage_zero_kernel<<<1, 256, 0, stream>>>(st, 2 * DH);
    sage_stats_kernel<<<512, 256, 0, stream>>>(buf0, st, N);
    sage_bn_kernel<<<bnGrid, 256, 0, stream>>>(buf0, st, g_h, be_h, feat, NF, invN, 1);

    // --- SAGE conv 1: BN(mean@Wl1 + bl1 + feat@Wr1) -> out1 ---
    sage_zero_kernel<<<ZB, 256, 0, stream>>>(buf1, NF);
    sage_zero_kernel<<<ZB, 256, 0, stream>>>(cnt, (size_t)N);
    sage_scatter_kernel<<<scatGrid, 256, 0, stream>>>(feat, e_src, e_dst, buf1, cnt, E);
    sage_meandiv_kernel<<<bnGrid, 256, 0, stream>>>(buf1, cnt, NF);
    sage_gemm_kernel<<<gemmGrid, 128, smemH, stream>>>(buf1, WT_l1, bl1, buf0, N, DH, 0);
    sage_gemm_kernel<<<gemmGrid, 128, smemH, stream>>>(feat, WT_r1, nullptr, buf0, N, DH, 1);
    sage_zero_kernel<<<1, 256, 0, stream>>>(st, 2 * DH);
    sage_stats_kernel<<<512, 256, 0, stream>>>(buf0, st, N);
    sage_bn_kernel<<<bnGrid, 256, 0, stream>>>(buf0, st, g1, be1, out1, NF, invN, 0);

    // --- SAGE conv 2: BN(mean@Wl2 + bl2 + out1@Wr2) -> out1 ---
    sage_zero_kernel<<<ZB, 256, 0, stream>>>(buf1, NF);
    sage_zero_kernel<<<ZB, 256, 0, stream>>>(cnt, (size_t)N);
    sage_scatter_kernel<<<scatGrid, 256, 0, stream>>>(out1, e_src, e_dst, buf1, cnt, E);
    sage_meandiv_kernel<<<bnGrid, 256, 0, stream>>>(buf1, cnt, NF);
    sage_gemm_kernel<<<gemmGrid, 128, smemH, stream>>>(buf1, WT_l2, bl2, buf0, N, DH, 0);
    sage_gemm_kernel<<<gemmGrid, 128, smemH, stream>>>(out1, WT_r2, nullptr, buf0, N, DH, 1);
    sage_zero_kernel<<<1, 256, 0, stream>>>(st, 2 * DH);
    sage_stats_kernel<<<512, 256, 0, stream>>>(buf0, st, N);
    sage_bn_kernel<<<bnGrid, 256, 0, stream>>>(buf0, st, g2, be2, out1, NF, invN, 0);
}
